// ViewLoRALinear_43319040147751
// MI455X (gfx1250) — matrix operand order repaired and checked by reading, not compiled, not run on hardware
//
#include <hip/hip_runtime.h>
#include <hip/hip_bf16.h>

typedef __attribute__((ext_vector_type(16))) __bf16 v16bf;
typedef __attribute__((ext_vector_type(8)))  float  v8f;

union Frag16 {          // 16 bf16 = one WMMA A/B operand (8 VGPRs)
    v16bf v;
    uint4 q[2];
};

#define TILE_M 128
#define TILE_N 128
#define KSTEP  32
#define LDA    40        // padded LDS row stride in bf16 elems (80B -> conflict-free b128 reads)

// ---------------------------------------------------------------------------
// Kernel 0: pack W[N][K] fp32 into B-operand layout, bf16 hi/lo split.
// Bpk[kb][n][j] = bf16(W[n][kb*16 + j]),  kb in [0, K/16)
// ---------------------------------------------------------------------------
__global__ void pack_w_kernel(const float* __restrict__ W,
                              __bf16* __restrict__ Whi,
                              __bf16* __restrict__ Wlo,
                              int N, int K)
{
    int t  = blockIdx.x * blockDim.x + threadIdx.x;   // one thread per (kb, n)
    int n  = t % N;
    int kb = t / N;
    if (kb >= (K >> 4)) return;

    const float* src = W + (size_t)n * K + kb * 16;
    __bf16 hbuf[16], lbuf[16];
#pragma unroll
    for (int j = 0; j < 16; ++j) {
        float f  = src[j];
        __bf16 h = (__bf16)f;
        hbuf[j]  = h;
        lbuf[j]  = (__bf16)(f - (float)h);
    }
    size_t o = ((size_t)kb * N + n) * 16;
    ((uint4*)(Whi + o))[0] = ((uint4*)hbuf)[0];
    ((uint4*)(Whi + o))[1] = ((uint4*)hbuf)[1];
    ((uint4*)(Wlo + o))[0] = ((uint4*)lbuf)[0];
    ((uint4*)(Wlo + o))[1] = ((uint4*)lbuf)[1];
}

// ---------------------------------------------------------------------------
// Kernel 1: h[m][r] = sum_k x[m][k] * loraA[view(m)][k][r],  R = 4.
// One wave32 per row; butterfly reduce.
// ---------------------------------------------------------------------------
__global__ void lora_h_kernel(const float* __restrict__ x,
                              const float* __restrict__ loraA,
                              const int*   __restrict__ view_idx,
                              float* __restrict__ h,
                              int M, int K, int T, int rep, int nviews)
{
    int row  = blockIdx.x * 8 + (threadIdx.x >> 5);
    int lane = threadIdx.x & 31;
    if (row >= M) return;

    int v = view_idx[(row / T) / rep];
    v = v < 0 ? 0 : (v >= nviews ? nviews - 1 : v);

    const float* xr = x + (size_t)row * K;
    const float* Ar = loraA + (size_t)v * K * 4;

    float a0 = 0.f, a1 = 0.f, a2 = 0.f, a3 = 0.f;
    for (int k = lane; k < K; k += 32) {
        float  xv = xr[k];
        float4 av = *(const float4*)(Ar + (size_t)k * 4);
        a0 += xv * av.x; a1 += xv * av.y; a2 += xv * av.z; a3 += xv * av.w;
    }
#pragma unroll
    for (int off = 16; off > 0; off >>= 1) {
        a0 += __shfl_xor(a0, off);
        a1 += __shfl_xor(a1, off);
        a2 += __shfl_xor(a2, off);
        a3 += __shfl_xor(a3, off);
    }
    if (lane == 0)
        ((float4*)h)[row] = make_float4(a0, a1, a2, a3);
}

// ---------------------------------------------------------------------------
// Kernel 2: split-precision bf16 WMMA GEMM + fused bias/LoRA epilogue.
// out[m][n] = sum_k x*W + b[n] + sum_r h[m][r]*loraB[view(m)][r][n]
// ---------------------------------------------------------------------------
__global__ __launch_bounds__(256)
void viewlora_gemm_kernel(const float* __restrict__ x,
                          const float* __restrict__ bias,
                          const __bf16* __restrict__ Whi,
                          const __bf16* __restrict__ Wlo,
                          const float* __restrict__ h,
                          const float* __restrict__ loraB,
                          const int*   __restrict__ view_idx,
                          float* __restrict__ out,
                          int K, int N, int T, int rep, int nviews)
{
    __shared__ __bf16 Ahi[TILE_M * LDA];
    __shared__ __bf16 Alo[TILE_M * LDA];

    const int tid  = threadIdx.x;
    const int wave = tid >> 5;         // 0..7
    const int lane = tid & 31;
    const int wm   = wave >> 2;        // 0..1  -> +64 rows
    const int wn   = wave & 3;         // 0..3  -> +32 cols
    const int g    = lane >> 4;        // half-wave group
    const int ln   = lane & 15;

    const int mtile = blockIdx.x * TILE_M;
    const int ntile = blockIdx.y * TILE_N;

    v8f acc[4][2];
#pragma unroll
    for (int i = 0; i < 4; ++i)
#pragma unroll
        for (int j = 0; j < 2; ++j)
            acc[i][j] = (v8f)0.0f;

    const int ldsRow = tid >> 1;            // 0..127
    const int ldsCol = (tid & 1) * 16;      // 0 or 16

    for (int ks = 0; ks < K; ks += KSTEP) {
        __syncthreads();
        // ---- stage x tile (128 x 32 fp32) into LDS with hi/lo bf16 split ----
        {
            const float* src = x + (size_t)(mtile + ldsRow) * K + ks + ldsCol;
            float vals[16];
            ((float4*)vals)[0] = ((const float4*)src)[0];
            ((float4*)vals)[1] = ((const float4*)src)[1];
            ((float4*)vals)[2] = ((const float4*)src)[2];
            ((float4*)vals)[3] = ((const float4*)src)[3];
            __bf16 hbuf[16], lbuf[16];
#pragma unroll
            for (int j = 0; j < 16; ++j) {
                float  f  = vals[j];
                __bf16 hv = (__bf16)f;
                hbuf[j]   = hv;
                lbuf[j]   = (__bf16)(f - (float)hv);
            }
            __bf16* dh = Ahi + ldsRow * LDA + ldsCol;
            __bf16* dl = Alo + ldsRow * LDA + ldsCol;
            ((uint4*)dh)[0] = ((uint4*)hbuf)[0];
            ((uint4*)dh)[1] = ((uint4*)hbuf)[1];
            ((uint4*)dl)[0] = ((uint4*)lbuf)[0];
            ((uint4*)dl)[1] = ((uint4*)lbuf)[1];
        }
        __syncthreads();

        // ---- B fragments straight from L2-resident packed W ----
        // lane group g covers K window [ks + g*16, ks + g*16 + 16), column = ln
        Frag16 bh[2], bl[2];
#pragma unroll
        for (int sn = 0; sn < 2; ++sn) {
            int    n = ntile + wn * 32 + sn * 16 + ln;
            size_t o = ((size_t)((ks >> 4) + g) * N + n) * 16;
            bh[sn].q[0] = ((const uint4*)(Whi + o))[0];
            bh[sn].q[1] = ((const uint4*)(Whi + o))[1];
            bl[sn].q[0] = ((const uint4*)(Wlo + o))[0];
            bl[sn].q[1] = ((const uint4*)(Wlo + o))[1];
        }

        // ---- A fragments from LDS, 3-term split accumulation ----
#pragma unroll
        for (int sm = 0; sm < 4; ++sm) {
            const int rbase = (wm * 64 + sm * 16 + ln) * LDA + g * 16;
            Frag16 ah, al;
            ah.q[0] = *(const uint4*)(Ahi + rbase);
            ah.q[1] = *(const uint4*)(Ahi + rbase + 8);
            al.q[0] = *(const uint4*)(Alo + rbase);
            al.q[1] = *(const uint4*)(Alo + rbase + 8);
#pragma unroll
            for (int sn = 0; sn < 2; ++sn) {
                acc[sm][sn] = __builtin_amdgcn_wmma_f32_16x16x32_bf16(
                    false, al.v, false, bh[sn].v, (short)0, acc[sm][sn], false, false);
                acc[sm][sn] = __builtin_amdgcn_wmma_f32_16x16x32_bf16(
                    false, ah.v, false, bl[sn].v, (short)0, acc[sm][sn], false, false);
                acc[sm][sn] = __builtin_amdgcn_wmma_f32_16x16x32_bf16(
                    false, ah.v, false, bh[sn].v, (short)0, acc[sm][sn], false, false);
            }
        }
    }

    // ---- epilogue: bias + rank-4 LoRA back-projection ----
#pragma unroll
    for (int sm = 0; sm < 4; ++sm) {
#pragma unroll
        for (int sn = 0; sn < 2; ++sn) {
            int   n  = ntile + wn * 32 + sn * 16 + ln;
            float bn = bias[n];
#pragma unroll
            for (int v = 0; v < 8; ++v) {
                int m  = mtile + wm * 64 + sm * 16 + g * 8 + v;
                int vw = view_idx[(m / T) / rep];
                vw = vw < 0 ? 0 : (vw >= nviews ? nviews - 1 : vw);
                float4 hv = *(const float4*)(h + (size_t)m * 4);
                const float* lb = loraB + (size_t)vw * 4 * N + n;
                float val = acc[sm][sn][v] + bn;
                val += hv.x * lb[0 * N] + hv.y * lb[1 * N]
                     + hv.z * lb[2 * N] + hv.w * lb[3 * N];
                out[(size_t)m * N + n] = val;
            }
        }
    }
}

// ---------------------------------------------------------------------------
extern "C" void kernel_launch(void* const* d_in, const int* in_sizes, int n_in,
                              void* d_out, int out_size, void* d_ws, size_t ws_size,
                              hipStream_t stream)
{
    const float* x        = (const float*)d_in[0];   // [BF, T, D_IN]
    const int*   view_idx = (const int*)  d_in[1];   // [B]
    const float* W        = (const float*)d_in[2];   // [D_OUT, D_IN]
    const float* b        = (const float*)d_in[3];   // [D_OUT]
    const float* loraA    = (const float*)d_in[4];   // [NV, D_IN, 4]
    const float* loraB    = (const float*)d_in[5];   // [NV, 4, D_OUT]

    const int D_IN   = 1024;
    const int D_OUT  = 1024;
    const int T      = 196;
    const int NV     = 11;
    const int Bsz    = in_sizes[1];                  // 16
    const int M      = in_sizes[0] / D_IN;           // BF*T = 25088
    const int BF     = M / T;                        // 128
    const int rep    = BF / Bsz;                     // 8

    // workspace: Whi(2MB) | Wlo(2MB) | h(M*4 floats)
    __bf16* Whi = (__bf16*)d_ws;
    __bf16* Wlo = Whi + (size_t)D_IN * D_OUT;
    float*  hbuf = (float*)((char*)d_ws + 2u * (size_t)D_IN * D_OUT * sizeof(__bf16));
    float*  out  = (float*)d_out;

    // 1) pack W into bf16 hi/lo B-operand layout
    {
        int total  = (D_IN >> 4) * D_OUT;            // 65536
        int blocks = (total + 255) / 256;
        pack_w_kernel<<<blocks, 256, 0, stream>>>(W, Whi, Wlo, D_OUT, D_IN);
    }
    // 2) low-rank projection h = x @ loraA[view]
    {
        int blocks = (M + 7) / 8;
        lora_h_kernel<<<blocks, 256, 0, stream>>>(x, loraA, view_idx, hbuf,
                                                  M, D_IN, T, rep, NV);
    }
    // 3) main WMMA GEMM with fused bias + LoRA epilogue
    {
        dim3 grid(M / TILE_M, D_OUT / TILE_N);       // (196, 8)
        viewlora_gemm_kernel<<<grid, 256, 0, stream>>>(
            x, b, Whi, Wlo, hbuf, loraB, view_idx, out,
            D_IN, D_OUT, T, rep, NV);
    }
}
